// CEA_10247791968685
// MI455X (gfx1250) — compile-verified, matrix operand order checked by reading, not run on hardware
//
#include <hip/hip_runtime.h>
#include <hip/hip_bf16.h>

typedef __attribute__((ext_vector_type(16))) _Float16 v16h;
typedef __attribute__((ext_vector_type(8)))  float    v8f;

#define DIMX   512
#define NSEQ   4096
#define BATCHN 4
#define HEADS  8
#define DHEAD  64
#define OCONV  1024
#define KCONV  1536
#define SCALEF 0.125f     // 64^-0.5

#define LDT 40            // LDS tile row stride (halfs): 80B rows keep 16B alignment for fragment gathers

#define WMMA_F16(a, b, c) \
  __builtin_amdgcn_wmma_f32_16x16x32_f16(false, (a), false, (b), (short)0, (c), false, false)

// Gather one 16x32 f16 fragment from an LDS tile (row-major, stride LDT).
// A-operand: rows = M. B-operand: tile is stored transposed [N][K], rows = N.
// Per CDNA5 ISA layout: lanes 0-15 hold K{0..7,16..23}, lanes 16-31 hold K{8..15,24..31}.
static __device__ __forceinline__ v16h frag_from_lds(const _Float16* tile, int row0, int lane) {
  const int r    = row0 + (lane & 15);
  const int kOff = (lane & 16) ? 8 : 0;
  const _Float16* p = tile + r * LDT + kOff;
  v16h f;
#pragma unroll
  for (int j = 0; j < 8; ++j) { f[j] = p[j]; f[j + 8] = p[j + 16]; }
  return f;
}

static __device__ __forceinline__ v8f zero8() {
  v8f z = {0.f, 0.f, 0.f, 0.f, 0.f, 0.f, 0.f, 0.f};
  return z;
}

// CDNA5 async copy: 32 bytes per lane, global -> LDS, tracked by ASYNCcnt.
// INST_OFFSET applies to both the global and LDS address (ISA 10.x async pseudocode).
static __device__ __forceinline__ void async_copy32(const _Float16* g, _Float16* l) {
  unsigned           lo = (unsigned)(uintptr_t)l;            // low 32 bits == LDS byte offset
  unsigned long long ga = (unsigned long long)(uintptr_t)g;
  asm volatile("global_load_async_to_lds_b128 %0, %1, off\n\t"
               "global_load_async_to_lds_b128 %0, %1, off offset:16"
               :: "v"(lo), "v"(ga) : "memory");
}

static __device__ __forceinline__ void wait_async0() {
  asm volatile("s_wait_asynccnt 0x0" ::: "memory");
}

// ---------------- elementwise / prep kernels ----------------

__global__ void k_cast_f16(const float* __restrict__ src, _Float16* __restrict__ dst, int n) {
  int i = blockIdx.x * blockDim.x + threadIdx.x;
  if (i < n) dst[i] = (_Float16)src[i];
}

// w_conv [oc=1024][ic=512][t=3] -> wc[(t*512+ic)][oc]  (GEMM-ready, row-major KxN)
__global__ void k_pack_conv_w(const float* __restrict__ w, _Float16* __restrict__ dst) {
  int i = blockIdx.x * blockDim.x + threadIdx.x;
  if (i >= OCONV * DIMX * 3) return;
  int oc  = i / (DIMX * 3);
  int rem = i - oc * (DIMX * 3);
  int ic  = rem / 3;
  int t   = rem - ic * 3;
  dst[(size_t)(t * DIMX + ic) * OCONV + oc] = (_Float16)w[i];
}

// column stats over sequence: max and sum(exp(k-max)) per (b, channel)
__global__ void k_kstat(const _Float16* __restrict__ km, float* __restrict__ kmax,
                        float* __restrict__ ksum) {
  int idx = blockIdx.x * blockDim.x + threadIdx.x;   // 0..B*512-1
  int b = idx >> 9;
  int c = idx & 511;
  const _Float16* p = km + (size_t)b * NSEQ * DIMX + c;
  float mx = -1e30f;
  for (int n = 0; n < NSEQ; ++n) {
    float v = (float)p[(size_t)n * DIMX];
    mx = fmaxf(mx, v);
  }
  float s = 0.f;
  for (int n = 0; n < NSEQ; ++n) {
    float v = (float)p[(size_t)n * DIMX];
    s += __expf(v - mx);
  }
  kmax[idx] = mx;
  ksum[idx] = s;
}

// in-place: k <- exp(k - colmax)   (unnormalized; 1/sum folded into ctx epilogue)
__global__ void k_expk(_Float16* __restrict__ km, const float* __restrict__ kmax, int total) {
  int i = blockIdx.x * blockDim.x + threadIdx.x;
  if (i >= total) return;
  int c = i & 511;
  int b = i / (NSEQ * DIMX);
  float m = kmax[b * DIMX + c];
  km[i] = (_Float16)__expf((float)km[i] - m);
}

// in-place softmax over each contiguous 64-wide head segment of q
__global__ void k_qsm(_Float16* __restrict__ q, int nseg) {
  int i = blockIdx.x * blockDim.x + threadIdx.x;
  if (i >= nseg) return;
  _Float16* p = q + (size_t)i * DHEAD;
  float mx = -1e30f;
  for (int j = 0; j < DHEAD; ++j) mx = fmaxf(mx, (float)p[j]);
  float s = 0.f;
  for (int j = 0; j < DHEAD; ++j) s += __expf((float)p[j] - mx);
  float inv = 1.0f / s;
  for (int j = 0; j < DHEAD; ++j) p[j] = (_Float16)(__expf((float)p[j] - mx) * inv);
}

// ---------------- fused conv(k=3,pad=1) + GLU via im2col WMMA GEMM ----------------
// qk[b,n,oc] = a * sigmoid(g), {a,g} = im2col(x)[n,1536] @ wc[1536,1024]
// Double-buffered: async A tiles (ASYNCcnt), ds-scatter B tiles; 1 barrier/K-step.
// grid: (8 oc-tiles of 64, 64 n-tiles of 64, B); block 128 (4 waves)
__global__ __launch_bounds__(128) void k_conv_glu(const _Float16* __restrict__ xh,
                                                  const _Float16* __restrict__ wc,
                                                  _Float16* __restrict__ qk) {
  __shared__ _Float16 As [2][64 * LDT];
  __shared__ _Float16 BaT[2][64 * LDT];
  __shared__ _Float16 BgT[2][64 * LDT];
  const int tid = threadIdx.x, lane = tid & 31, wave = tid >> 5;
  const int b   = blockIdx.z;
  const int n0  = blockIdx.y * 64;
  const int oc0 = blockIdx.x * 64;
  const int wm  = (wave >> 1) * 32, wn = (wave & 1) * 32;

  const int rA = tid >> 1, segA = (tid & 1) * 16;     // A stage: 64 rows x 32 halfs
  const int kkB = tid >> 2, csB = (tid & 3) * 16;     // B stage: 32 rows x 64 cols

  v8f acc_a[4], acc_g[4];
#pragma unroll
  for (int f = 0; f < 4; ++f) { acc_a[f] = zero8(); acc_g[f] = zero8(); }

  auto stageA = [&](int k0, int buf) {
    const int t   = k0 >> 9;            // conv tap; k-tile never straddles taps (512%32==0)
    const int ic0 = k0 - (t << 9);
    const int seq = n0 + rA + t - 1;
    _Float16* dst = &As[buf][rA * LDT + segA];
    if (seq >= 0 && seq < NSEQ) {
      async_copy32(xh + ((size_t)b * NSEQ + seq) * DIMX + ic0 + segA, dst);
    } else {
      uint4 z = {0u, 0u, 0u, 0u};
      ((uint4*)dst)[0] = z;
      ((uint4*)dst)[1] = z;
    }
  };
  auto loadB = [&](int k0, _Float16* ta, _Float16* tg) {
    const _Float16* rp = wc + (size_t)(k0 + kkB) * OCONV + oc0 + csB;
    *(uint4*)(ta)     = *(const uint4*)(rp);
    *(uint4*)(ta + 8) = *(const uint4*)(rp + 8);
    *(uint4*)(tg)     = *(const uint4*)(rp + 512);
    *(uint4*)(tg + 8) = *(const uint4*)(rp + 520);
  };
  auto scatterB = [&](int buf, const _Float16* ta, const _Float16* tg) {
#pragma unroll
    for (int j = 0; j < 16; ++j) {
      BaT[buf][(csB + j) * LDT + kkB] = ta[j];
      BgT[buf][(csB + j) * LDT + kkB] = tg[j];
    }
  };

  // prologue: stage tile 0 into buffer 0
  stageA(0, 0);
  {
    _Float16 ta[16], tg[16];
    loadB(0, ta, tg);
    scatterB(0, ta, tg);
  }

  for (int k0 = 0, it = 0; k0 < KCONV; k0 += 32, ++it) {
    const int cur = it & 1;
    wait_async0();
    __syncthreads();
    const bool hasNext = (k0 + 32) < KCONV;
    _Float16 ta[16], tg[16];
    if (hasNext) {
      stageA(k0 + 32, cur ^ 1);
      loadB(k0 + 32, ta, tg);
    }
    v16h a0  = frag_from_lds(As[cur],  wm,      lane);
    v16h a1  = frag_from_lds(As[cur],  wm + 16, lane);
    v16h ba0 = frag_from_lds(BaT[cur], wn,      lane);
    v16h ba1 = frag_from_lds(BaT[cur], wn + 16, lane);
    v16h bg0 = frag_from_lds(BgT[cur], wn,      lane);
    v16h bg1 = frag_from_lds(BgT[cur], wn + 16, lane);
    acc_a[0] = WMMA_F16(a0, ba0, acc_a[0]);
    acc_a[1] = WMMA_F16(a0, ba1, acc_a[1]);
    acc_a[2] = WMMA_F16(a1, ba0, acc_a[2]);
    acc_a[3] = WMMA_F16(a1, ba1, acc_a[3]);
    acc_g[0] = WMMA_F16(a0, bg0, acc_g[0]);
    acc_g[1] = WMMA_F16(a0, bg1, acc_g[1]);
    acc_g[2] = WMMA_F16(a1, bg0, acc_g[2]);
    acc_g[3] = WMMA_F16(a1, bg1, acc_g[3]);
    if (hasNext) scatterB(cur ^ 1, ta, tg);
  }

  // GLU epilogue (C layout: VGPR r -> M = r (+8 for lanes 16-31), N = lane&15)
  const int mhi = (lane & 16) ? 8 : 0, nlo = lane & 15;
#pragma unroll
  for (int mi = 0; mi < 2; ++mi)
#pragma unroll
    for (int ni = 0; ni < 2; ++ni) {
      int f = mi * 2 + ni;
#pragma unroll
      for (int r = 0; r < 8; ++r) {
        int m = n0 + wm + mi * 16 + r + mhi;
        int n = oc0 + wn + ni * 16 + nlo;
        float av = acc_a[f][r], gv = acc_g[f][r];
        qk[((size_t)b * NSEQ + m) * DIMX + n] = (_Float16)(av / (1.0f + __expf(-gv)));
      }
    }
}

// ---------------- generic batched f16 WMMA GEMM (double-buffered, async A) ----------------
// C[m,n] = sum_k A[m,k]*B[k,n]; per-z offsets: base + (z/hdiv)*s?1 + (z%hdiv)*s?2
// out_f32: store float with optional bias, else store f16.
__global__ __launch_bounds__(128) void k_gemm(const _Float16* __restrict__ A,
                                              const _Float16* __restrict__ Bm,
                                              void* __restrict__ Cv,
                                              int K, int lda, int ldb, int ldc, int hdiv,
                                              long long sA1, long long sA2,
                                              long long sB1, long long sB2,
                                              long long sC1, long long sC2,
                                              const float* __restrict__ bias, int out_f32) {
  __shared__ _Float16 As [2][64 * LDT];
  __shared__ _Float16 BsT[2][64 * LDT];
  const int tid = threadIdx.x, lane = tid & 31, wave = tid >> 5;
  const int z  = blockIdx.z;
  const int zb = z / hdiv, zh = z - zb * hdiv;
  A  += zb * sA1 + zh * sA2;
  Bm += zb * sB1 + zh * sB2;
  const int m0 = blockIdx.y * 64;
  const int n0 = blockIdx.x * 64;
  const int wm = (wave >> 1) * 32, wn = (wave & 1) * 32;

  const int rA = tid >> 1, segA = (tid & 1) * 16;
  const int kkB = tid >> 2, csB = (tid & 3) * 16;

  v8f acc[4];
#pragma unroll
  for (int f = 0; f < 4; ++f) acc[f] = zero8();

  // prologue: stage tile 0 into buffer 0
  async_copy32(A + (size_t)(m0 + rA) * lda + segA, &As[0][rA * LDT + segA]);
  {
    const _Float16* rp = Bm + (size_t)kkB * ldb + n0 + csB;
    _Float16 tb[16];
    *(uint4*)(tb)     = *(const uint4*)(rp);
    *(uint4*)(tb + 8) = *(const uint4*)(rp + 8);
#pragma unroll
    for (int j = 0; j < 16; ++j) BsT[0][(csB + j) * LDT + kkB] = tb[j];
  }

  for (int k0 = 0, it = 0; k0 < K; k0 += 32, ++it) {
    const int cur = it & 1;
    wait_async0();
    __syncthreads();
    const bool hasNext = (k0 + 32) < K;
    _Float16 tb[16];
    if (hasNext) {
      async_copy32(A + (size_t)(m0 + rA) * lda + (k0 + 32) + segA,
                   &As[cur ^ 1][rA * LDT + segA]);
      const _Float16* rp = Bm + (size_t)(k0 + 32 + kkB) * ldb + n0 + csB;
      *(uint4*)(tb)     = *(const uint4*)(rp);
      *(uint4*)(tb + 8) = *(const uint4*)(rp + 8);
      __builtin_prefetch(Bm + (size_t)(k0 + 64 + kkB) * ldb + n0 + csB, 0, 3);
    }
    v16h a0 = frag_from_lds(As[cur],  wm,      lane);
    v16h a1 = frag_from_lds(As[cur],  wm + 16, lane);
    v16h b0 = frag_from_lds(BsT[cur], wn,      lane);
    v16h b1 = frag_from_lds(BsT[cur], wn + 16, lane);
    acc[0] = WMMA_F16(a0, b0, acc[0]);
    acc[1] = WMMA_F16(a0, b1, acc[1]);
    acc[2] = WMMA_F16(a1, b0, acc[2]);
    acc[3] = WMMA_F16(a1, b1, acc[3]);
    if (hasNext) {
#pragma unroll
      for (int j = 0; j < 16; ++j) BsT[cur ^ 1][(csB + j) * LDT + kkB] = tb[j];
    }
  }

  const int mhi = (lane & 16) ? 8 : 0, nlo = lane & 15;
#pragma unroll
  for (int mi = 0; mi < 2; ++mi)
#pragma unroll
    for (int ni = 0; ni < 2; ++ni) {
      int f = mi * 2 + ni;
#pragma unroll
      for (int r = 0; r < 8; ++r) {
        int m = m0 + wm + mi * 16 + r + mhi;
        int n = n0 + wn + ni * 16 + nlo;
        float v = acc[f][r];
        if (out_f32) {
          float* C = (float*)Cv + zb * sC1 + zh * sC2;
          C[(size_t)m * ldc + n] = v + (bias ? bias[n] : 0.0f);
        } else {
          _Float16* C = (_Float16*)Cv + zb * sC1 + zh * sC2;
          C[(size_t)m * ldc + n] = (_Float16)v;
        }
      }
    }
}

// ---------------- context GEMM: ctx[b,h,d,e] = SCALE/ksum[d] * sum_n e[n,d]*v[n,e] ----
// A is the transpose of the stored e-matrix; transpose happens on the LDS write
// (both operands transposed -> stays on the ds-scatter path, no async).
// grid: 32 blocks (b*8+h); block 128; K loop over N=4096.
__global__ __launch_bounds__(128) void k_ctx(const _Float16* __restrict__ em,
                                             const _Float16* __restrict__ vm,
                                             const float* __restrict__ ksum,
                                             _Float16* __restrict__ ctx) {
  __shared__ _Float16 As [64 * LDT];   // [d][k]
  __shared__ _Float16 BsT[64 * LDT];   // [e][k]
  const int tid = threadIdx.x, lane = tid & 31, wave = tid >> 5;
  const int b = blockIdx.x >> 3, h = blockIdx.x & 7;
  const int wm = (wave >> 1) * 32, wn = (wave & 1) * 32;
  const size_t baseRow = (size_t)b * NSEQ;
  const int hc = h * DHEAD;

  v8f acc[4];
#pragma unroll
  for (int f = 0; f < 4; ++f) acc[f] = zero8();

  const int kk = tid & 31, d0 = (tid >> 5) * 16;
  for (int k0 = 0; k0 < NSEQ; k0 += 32) {
    {
      const uint4* sp = (const uint4*)(em + (baseRow + k0 + kk) * DIMX + hc + d0);
      _Float16 tmp[16];
      *(uint4*)(tmp) = sp[0]; *(uint4*)(tmp + 8) = sp[1];
#pragma unroll
      for (int j = 0; j < 16; ++j) As[(d0 + j) * LDT + kk] = tmp[j];
    }
    {
      const uint4* sp = (const uint4*)(vm + (baseRow + k0 + kk) * DIMX + hc + d0);
      _Float16 tmp[16];
      *(uint4*)(tmp) = sp[0]; *(uint4*)(tmp + 8) = sp[1];
#pragma unroll
      for (int j = 0; j < 16; ++j) BsT[(d0 + j) * LDT + kk] = tmp[j];
    }
    __syncthreads();
    v16h a0 = frag_from_lds(As,  wm,      lane);
    v16h a1 = frag_from_lds(As,  wm + 16, lane);
    v16h b0 = frag_from_lds(BsT, wn,      lane);
    v16h b1 = frag_from_lds(BsT, wn + 16, lane);
    acc[0] = WMMA_F16(a0, b0, acc[0]);
    acc[1] = WMMA_F16(a0, b1, acc[1]);
    acc[2] = WMMA_F16(a1, b0, acc[2]);
    acc[3] = WMMA_F16(a1, b1, acc[3]);
    __syncthreads();
  }

  const int mhi = (lane & 16) ? 8 : 0, nlo = lane & 15;
#pragma unroll
  for (int mi = 0; mi < 2; ++mi)
#pragma unroll
    for (int ni = 0; ni < 2; ++ni) {
      int f = mi * 2 + ni;
#pragma unroll
      for (int r = 0; r < 8; ++r) {
        int d = wm + mi * 16 + r + mhi;
        int e = wn + ni * 16 + nlo;
        float s = SCALEF / ksum[b * DIMX + hc + d];
        ctx[((size_t)(b * HEADS + h) * DHEAD + d) * DHEAD + e] = (_Float16)(acc[f][r] * s);
      }
    }
}

// ---------------- host orchestration ----------------

extern "C" void kernel_launch(void* const* d_in, const int* in_sizes, int n_in,
                              void* d_out, int out_size, void* d_ws, size_t ws_size,
                              hipStream_t stream) {
  const float* x      = (const float*)d_in[0];
  const float* w_conv = (const float*)d_in[1];
  const float* wq     = (const float*)d_in[2];
  const float* wk     = (const float*)d_in[3];
  const float* wv     = (const float*)d_in[4];
  const float* wo     = (const float*)d_in[5];
  const float* bo     = (const float*)d_in[6];
  float* out = (float*)d_out;

  const size_t XN = (size_t)BATCHN * NSEQ * DIMX;   // 8388608
  const size_t WN = (size_t)DIMX * DIMX;            // 262144

  char* p = (char*)d_ws;
  auto alloc = [&](size_t bytes) {
    char* r = p;
    p += (bytes + 255) & ~(size_t)255;
    return r;
  };
  _Float16* xh   = (_Float16*)alloc(XN * 2);
  _Float16* qkh  = (_Float16*)alloc(XN * 2);   // reused for attn output
  _Float16* qh   = (_Float16*)alloc(XN * 2);
  _Float16* kh   = (_Float16*)alloc(XN * 2);   // becomes exp(k-max) in place
  _Float16* vh   = (_Float16*)alloc(XN * 2);
  _Float16* wch  = (_Float16*)alloc((size_t)KCONV * OCONV * 2);
  _Float16* wqh  = (_Float16*)alloc(WN * 2);
  _Float16* wkh  = (_Float16*)alloc(WN * 2);
  _Float16* wvh  = (_Float16*)alloc(WN * 2);
  _Float16* woh  = (_Float16*)alloc(WN * 2);
  _Float16* ctxh = (_Float16*)alloc((size_t)BATCHN * HEADS * DHEAD * DHEAD * 2);
  float*    kmax = (float*)alloc(BATCHN * DIMX * 4);
  float*    ksum = (float*)alloc(BATCHN * DIMX * 4);

  // 1) precision casts / weight packing
  k_cast_f16<<<(int)((XN + 255) / 256), 256, 0, stream>>>(x, xh, (int)XN);
  k_cast_f16<<<(int)((WN + 255) / 256), 256, 0, stream>>>(wq, wqh, (int)WN);
  k_cast_f16<<<(int)((WN + 255) / 256), 256, 0, stream>>>(wk, wkh, (int)WN);
  k_cast_f16<<<(int)((WN + 255) / 256), 256, 0, stream>>>(wv, wvh, (int)WN);
  k_cast_f16<<<(int)((WN + 255) / 256), 256, 0, stream>>>(wo, woh, (int)WN);
  k_pack_conv_w<<<(OCONV * DIMX * 3 + 255) / 256, 256, 0, stream>>>(w_conv, wch);

  // 2) conv + GLU (im2col WMMA)
  dim3 gConv(OCONV / 2 / 64, NSEQ / 64, BATCHN);   // (8, 64, 4)
  k_conv_glu<<<gConv, 128, 0, stream>>>(xh, wch, qkh);

  // 3) q/k/v projections
  dim3 gProj(DIMX / 64, (BATCHN * NSEQ) / 64, 1);  // (8, 256, 1)
  k_gemm<<<gProj, 128, 0, stream>>>(qkh, wqh, qh, DIMX, DIMX, DIMX, DIMX, 1,
                                    0, 0, 0, 0, 0, 0, nullptr, 0);
  k_gemm<<<gProj, 128, 0, stream>>>(qkh, wkh, kh, DIMX, DIMX, DIMX, DIMX, 1,
                                    0, 0, 0, 0, 0, 0, nullptr, 0);
  k_gemm<<<gProj, 128, 0, stream>>>(xh, wvh, vh, DIMX, DIMX, DIMX, DIMX, 1,
                                    0, 0, 0, 0, 0, 0, nullptr, 0);

  // 4) k column-softmax stats, then k <- exp(k-max) in place
  k_kstat<<<(BATCHN * DIMX) / 256, 256, 0, stream>>>(kh, kmax, ksum);
  k_expk<<<(int)((XN + 255) / 256), 256, 0, stream>>>(kh, kmax, (int)XN);

  // 5) ctx = (1/sum)*SCALE * e^T v   per (b,h)
  k_ctx<<<BATCHN * HEADS, 128, 0, stream>>>(kh, vh, ksum, ctxh);

  // 6) q row-softmax over each head segment (in place)
  k_qsm<<<(BATCHN * NSEQ * HEADS + 255) / 256, 256, 0, stream>>>(qh, BATCHN * NSEQ * HEADS);

  // 7) attn = q_sm @ ctx   per (b,h); writes into qkh (free by now)
  dim3 gAttn(1, NSEQ / 64, BATCHN * HEADS);
  k_gemm<<<gAttn, 128, 0, stream>>>(
      qh, ctxh, qkh, DHEAD, DIMX, DHEAD, DIMX, HEADS,
      (long long)NSEQ * DIMX, (long long)DHEAD,                 // A: b-stride, h-stride
      (long long)HEADS * DHEAD * DHEAD, (long long)DHEAD * DHEAD,
      (long long)NSEQ * DIMX, (long long)DHEAD,
      nullptr, 0);

  // 8) output projection + bias -> f32 d_out
  k_gemm<<<gProj, 128, 0, stream>>>(qkh, woh, out, DIMX, DIMX, DIMX, DIMX, 1,
                                    0, 0, 0, 0, 0, 0, bo, 1);
}
